// MeanNegArcCos_Loss_69320772158194
// MI455X (gfx1250) — matrix operand order repaired and checked by reading, not compiled, not run on hardware
//
#include <hip/hip_runtime.h>

// ---------------------------------------------------------------------------
// MeanNegArcCos SupCon loss on MI455X (gfx1250)
//   K1: row L2-normalize fp32 -> f16 workspace
//   K2: Gram S = N * N^T via v_wmma_f32_16x16x32_f16, 64x64 tile per wave
//       (4x4 WMMA register tiling: 16 wmma per 8 fragment loads per K-step,
//        halves L2 fragment traffic vs 2x2 tiling)
//   K3: per-row masked softmax reduction (one block per row, S read once)
//   K4: final mean
// ---------------------------------------------------------------------------

#define BN 4096
#define DK 1024

constexpr float TEMP  = 0.07f;
constexpr float COS_M = 0.95533648912560601964f;  // cos(0.3)
constexpr float SIN_M = 0.29552020666133957511f;  // sin(0.3)

typedef __attribute__((ext_vector_type(16))) _Float16 v16h;
typedef __attribute__((ext_vector_type(8)))  _Float16 v8h;
typedef __attribute__((ext_vector_type(8)))  float    v8f;

// ---------------- K1: normalize rows, emit f16 ----------------
__global__ void k_normalize(const float* __restrict__ X, _Float16* __restrict__ Nh) {
    __shared__ float red[256];
    const int row = blockIdx.x;
    const int t   = threadIdx.x;
    const float* xr = X + (size_t)row * DK;
    float v[4];
    float ss = 0.0f;
#pragma unroll
    for (int c = 0; c < 4; ++c) {
        v[c] = xr[t + 256 * c];
        ss += v[c] * v[c];
    }
    red[t] = ss;
    __syncthreads();
    for (int s = 128; s > 0; s >>= 1) {
        if (t < s) red[t] += red[t + s];
        __syncthreads();
    }
    const float rn = rsqrtf(red[0]);
    _Float16* orow = Nh + (size_t)row * DK;
#pragma unroll
    for (int c = 0; c < 4; ++c) orow[t + 256 * c] = (_Float16)(v[c] * rn);
}

// ---------------- K2: WMMA Gram matrix ----------------
// A fragment (16x32 f16): lane m (0-15) = row m, holds K = {kh..kh+7} in v0..v3
// halves and K = {kh+16..kh+23} in v4..v7, kh = (lane>=16)*8.
__device__ inline v16h load_afrag(const _Float16* __restrict__ p, int k) {
    v8h lo = *(const v8h*)(p + k);
    v8h hi = *(const v8h*)(p + k + 16);
    v16h r;
#pragma unroll
    for (int e = 0; e < 8; ++e) { r[e] = lo[e]; r[e + 8] = hi[e]; }
    return r;
}
// B fragment (32x16 f16, column-major): lane n (0-15) = column n, holds K =
// {kh..kh+7} and {kh+16..kh+23}, kh = (lane>=16)*8. Column n of B == row n of N.
__device__ inline v16h load_bfrag(const _Float16* __restrict__ p, int k) {
    v8h lo = *(const v8h*)(p + k);
    v8h hi = *(const v8h*)(p + k + 16);
    v16h r;
#pragma unroll
    for (int e = 0; e < 8; ++e) { r[e] = lo[e]; r[e + 8] = hi[e]; }
    return r;
}

__global__ void __launch_bounds__(32)
k_gram_wmma(const _Float16* __restrict__ Nh, float* __restrict__ S) {
    const int lane = threadIdx.x;       // 0..31, wave32, EXEC all-ones
    const int rm = blockIdx.x * 64;     // output row base (64-wide row band)
    const int cn = blockIdx.y * 64;     // output col base (64-wide col band)
    const int lmod = lane & 15;
    const int lhi  = lane >> 4;
    const int ka = lhi * 8;             // A per-lane K chunk offset
    const int kb = lhi * 8;             // B per-lane K chunk offset

    const _Float16* arow[4];
    const _Float16* brow[4];
#pragma unroll
    for (int i = 0; i < 4; ++i) {
        arow[i] = Nh + (size_t)(rm + 16 * i + lmod) * DK;
        brow[i] = Nh + (size_t)(cn + 16 * i + lmod) * DK;
    }

    v8f acc[4][4];
#pragma unroll
    for (int i = 0; i < 4; ++i)
#pragma unroll
        for (int j = 0; j < 4; ++j) acc[i][j] = (v8f){};

    for (int k0 = 0; k0 < DK; k0 += 32) {
        v16h a[4], b[4];
#pragma unroll
        for (int i = 0; i < 4; ++i) a[i] = load_afrag(arow[i], k0 + ka);
#pragma unroll
        for (int j = 0; j < 4; ++j) b[j] = load_bfrag(brow[j], k0 + kb);
#pragma unroll
        for (int i = 0; i < 4; ++i)
#pragma unroll
            for (int j = 0; j < 4; ++j)
                acc[i][j] = __builtin_amdgcn_wmma_f32_16x16x32_f16(
                    false, a[i], false, b[j], (short)0, acc[i][j], false, false);
    }

    // C/D layout: VGPR d, lanes 0-15 -> M=d, N=lane; lanes 16-31 -> M=d+8, N=lane-16
    const int mo = lhi * 8;
#pragma unroll
    for (int i = 0; i < 4; ++i) {
#pragma unroll
        for (int j = 0; j < 4; ++j) {
            float* __restrict__ tile = S + (size_t)(rm + 16 * i + mo) * BN + (cn + 16 * j + lmod);
#pragma unroll
            for (int d = 0; d < 8; ++d) tile[(size_t)d * BN] = acc[i][j][d];
        }
    }
}

// ---------------- K3: per-row masked softmax reduction ----------------
__global__ void k_rowloss(const float* __restrict__ S, const int* __restrict__ labels,
                          float* __restrict__ rowloss) {
    __shared__ float  red[256];
    __shared__ float4 r4[256];
    const int i = blockIdx.x;
    const int t = threadIdx.x;
    const int li = labels[i];
    const float* srow = S + (size_t)i * BN;

    float comb[16];
    unsigned posmask = 0u, diagmask = 0u;
    float lmax = -1e30f;
#pragma unroll
    for (int c = 0; c < 16; ++c) {
        const int j = t + 256 * c;
        float s = srow[j];
        s = fminf(fmaxf(s, -1.0f + 1e-7f), 1.0f - 1e-7f);
        const float sn = sqrtf(fmaxf(1.0f - s * s, 0.0f));
        const float cm = s * COS_M - sn * SIN_M;       // cos(theta + m)
        const bool diag = (j == i);
        const bool pos  = (!diag) && (labels[j] == li);
        float lg;
        if (diag)      lg = 0.0f;                      // both masks zero -> 0
        else if (pos)  lg = -cm / TEMP;
        else           lg = -s / TEMP;
        comb[c] = lg;
        if (pos)  posmask  |= (1u << c);
        if (diag) diagmask |= (1u << c);
        lmax = fmaxf(lmax, lg);
    }

    red[t] = lmax;
    __syncthreads();
    for (int s2 = 128; s2 > 0; s2 >>= 1) {
        if (t < s2) red[t] = fmaxf(red[t], red[t + s2]);
        __syncthreads();
    }
    lmax = red[0];
    __syncthreads();

    float spe = 0.0f, sne = 0.0f, spl = 0.0f, pc = 0.0f;
#pragma unroll
    for (int c = 0; c < 16; ++c) {
        if (diagmask & (1u << c)) continue;
        const float ls = comb[c] - lmax;
        const float e  = expf(ls);
        if (posmask & (1u << c)) { spe += e; spl += ls; pc += 1.0f; }
        else                     { sne += e; }
    }
    r4[t] = make_float4(spe, sne, spl, pc);
    __syncthreads();
    for (int s2 = 128; s2 > 0; s2 >>= 1) {
        if (t < s2) {
            r4[t].x += r4[t + s2].x;
            r4[t].y += r4[t + s2].y;
            r4[t].z += r4[t + s2].z;
            r4[t].w += r4[t + s2].w;
        }
        __syncthreads();
    }
    if (t == 0) {
        const float pos_count = r4[0].w;
        const float neg_count = (float)(BN - 1) - pos_count;
        const float mean_exp_neg = r4[0].y / (neg_count + 1e-8f);
        const float denom = r4[0].x + mean_exp_neg;
        const float num = r4[0].z - pos_count * logf(denom);
        const float div = (pos_count < 1e-6f) ? 1.0f : pos_count;
        rowloss[i] = num / div;   // (TEMP/BASE_TEMPERATURE) == 1
    }
}

// ---------------- K4: mean over rows ----------------
__global__ void k_finalize(const float* __restrict__ rowloss, float* __restrict__ out) {
    __shared__ float red[256];
    const int t = threadIdx.x;
    float s = 0.0f;
#pragma unroll
    for (int c = 0; c < 16; ++c) s += rowloss[t + 256 * c];
    red[t] = s;
    __syncthreads();
    for (int k = 128; k > 0; k >>= 1) {
        if (t < k) red[t] += red[t + k];
        __syncthreads();
    }
    if (t == 0) out[0] = red[0] / (float)BN;
}

// ---------------------------------------------------------------------------
extern "C" void kernel_launch(void* const* d_in, const int* in_sizes, int n_in,
                              void* d_out, int out_size, void* d_ws, size_t ws_size,
                              hipStream_t stream) {
    const float* features = (const float*)d_in[0];
    const int*   labels   = (const int*)d_in[1];
    // d_in[2] (sigma) unused by the reference module.

    char* ws = (char*)d_ws;
    _Float16* Nh = (_Float16*)ws;                                   // 8 MB
    float* S = (float*)(ws + (size_t)BN * DK * sizeof(_Float16));   // 64 MB
    float* rowloss = (float*)(ws + (size_t)BN * DK * sizeof(_Float16)
                                 + (size_t)BN * BN * sizeof(float)); // 16 KB

    k_normalize<<<BN, 256, 0, stream>>>(features, Nh);
    dim3 grid(BN / 64, BN / 64);
    k_gram_wmma<<<grid, 32, 0, stream>>>(Nh, S);
    k_rowloss<<<BN, 256, 0, stream>>>(S, labels, rowloss);
    k_finalize<<<1, 256, 0, stream>>>(rowloss, (float*)d_out);
}
